// GlobalPointer_50431505990307
// MI455X (gfx1250) — matrix operand order repaired and checked by reading, not compiled, not run on hardware
//
#include <hip/hip_runtime.h>
#include <hip/hip_bf16.h>
#include <math.h>

typedef __attribute__((ext_vector_type(16))) _Float16 v16h;
typedef __attribute__((ext_vector_type(8)))  _Float16 v8h;
typedef __attribute__((ext_vector_type(4)))  _Float16 v4h;
typedef __attribute__((ext_vector_type(8)))  float    v8f;
typedef __attribute__((ext_vector_type(4)))  float    v4f;

#define HEADS   12
#define HSIZE   64
#define HIDDEN  768
#define OUTDIM  1536          // HEADS*2*HSIZE
#define NB      8
#define SEQ     512
#define NTOK    (NB*SEQ)      // 4096
#define NPAIR   32            // HSIZE/2 rope pairs
#define NEGINF  (-3.4028234663852886e+38f)

static __device__ __forceinline__ v16h cat16(v8h lo, v8h hi) {
  return __builtin_shufflevector(lo, hi, 0,1,2,3,4,5,6,7,8,9,10,11,12,13,14,15);
}

// ---------- kernel 0a: W (768x1536 f32, row-major) -> Wt (1536x768 f16, row-major)
__global__ void pack_w(const float* __restrict__ W, _Float16* __restrict__ Wt) {
  int idx = blockIdx.x * blockDim.x + threadIdx.x;     // n*HIDDEN + k
  if (idx >= OUTDIM * HIDDEN) return;
  int n = idx / HIDDEN;
  int k = idx - n * HIDDEN;
  Wt[idx] = (_Float16)W[k * OUTDIM + n];
}

// ---------- kernel 0b: X f32 -> f16 (same layout), 4 elems/thread
__global__ void pack_x(const float* __restrict__ X, _Float16* __restrict__ Xh) {
  int i = (blockIdx.x * blockDim.x + threadIdx.x) * 4;
  if (i >= NTOK * HIDDEN) return;
  v4f f = *(const v4f*)(X + i);
  v4h h = { (_Float16)f[0], (_Float16)f[1], (_Float16)f[2], (_Float16)f[3] };
  *(v4h*)(Xh + i) = h;
}

// ---------- kernel 0c: rope tables, layout [p][l]  (p=0..31, l=0..511)
__global__ void rope_tables(float* __restrict__ sintab, float* __restrict__ costab) {
  int idx = blockIdx.x * blockDim.x + threadIdx.x;     // p*SEQ + l
  if (idx >= NPAIR * SEQ) return;
  int p = idx >> 9;
  int l = idx & (SEQ - 1);
  float invf = exp2f((float)p * -0.415241011861f);     // 10000^(-p/32)
  float s, c;
  sincosf((float)l * invf, &s, &c);
  sintab[idx] = s;
  costab[idx] = c;
}

// ---------- kernel 1: x = Xh@Wt + b, RoPE, split q/k, store f16 [b][h][l][64]
__global__ __launch_bounds__(256) void qk_proj_rope(
    const _Float16* __restrict__ Xh,    // (4096, 768) f16
    const _Float16* __restrict__ Wt,    // (1536, 768) f16 (W transposed)
    const float*    __restrict__ bias,  // (1536,)
    const float*    __restrict__ sintab,// (32, 512)
    const float*    __restrict__ costab,// (32, 512)
    _Float16* __restrict__ qbuf,        // (8*12*512*64) f16
    _Float16* __restrict__ kbuf) {
  const int wave = threadIdx.x >> 5;
  const int lane = threadIdx.x & 31;
  const int tile = blockIdx.x * 8 + wave;     // 0 .. 256*96-1
  const int mt = tile / (OUTDIM / 16);        // token-tile  (0..255)
  const int nt = tile - mt * (OUTDIM / 16);   // chan-tile   (0..95)
  const int j  = lane & 15;
  const int hi = lane >> 4;

  // A operand: lane holds row m0+j; K runs [8*hi .. +7] and [+16 .. +23] per chunk
  const _Float16* arow = Xh + (mt * 16 + j) * HIDDEN + 8 * hi;
  // B operand: lane holds column nt*16+j; K runs [8*hi .. +7] and [+16 .. +23] per chunk
  const _Float16* brow = Wt + (nt * 16 + j) * HIDDEN + 8 * hi;

  v8f acc = {};
  #pragma unroll 4
  for (int kc = 0; kc < HIDDEN / 32; ++kc) {
    v8h a0 = *(const v8h*)(arow + kc * 32);
    v8h a1 = *(const v8h*)(arow + kc * 32 + 16);
    v8h b0 = *(const v8h*)(brow + kc * 32);
    v8h b1 = *(const v8h*)(brow + kc * 32 + 16);
    acc = __builtin_amdgcn_wmma_f32_16x16x32_f16(
        false, cat16(a0, a1), false, cat16(b0, b1),
        (short)0, acc, false, false);
  }

  // channel decode (uniform per wave except low 4 bits via j)
  const int cg = nt * 16 + j;          // global output channel
  const int h  = cg >> 7;              // head
  const int qk = (cg >> 6) & 1;        // 0=q, 1=k
  const int d  = cg & 63;              // dim within head
  const int p  = d >> 1;               // rope pair index
  const float bv = bias[cg];

  const int m0   = mt * 16;
  const int bidx = m0 >> 9;                    // batch (tile never crosses seq)
  const int l0   = (m0 & (SEQ - 1)) + 8 * hi;  // base token for this lane half

  // 8 consecutive l values per lane -> two b128 loads per table
  const float* sp = sintab + p * SEQ + l0;
  const float* cp = costab + p * SEQ + l0;
  v4f s01 = *(const v4f*)(sp);
  v4f s23 = *(const v4f*)(sp + 4);
  v4f c01 = *(const v4f*)(cp);
  v4f c23 = *(const v4f*)(cp + 4);

  _Float16* dst = (qk ? kbuf : qbuf) + ((bidx * HEADS + h) * SEQ) * HSIZE + d;

  #pragma unroll
  for (int r = 0; r < 8; ++r) {
    float v = acc[r] + bv;
    float w = __shfl_xor(v, 1, 32);            // partner channel d^1 (same rows)
    float s = (r < 4) ? s01[r & 3] : s23[r & 3];
    float c = (r < 4) ? c01[r & 3] : c23[r & 3];
    float o = (j & 1) ? (w * s + v * c)        // odd d:  x1*sin + x2*cos
                      : (v * c - w * s);       // even d: x1*cos - x2*sin
    dst[(l0 + r) * HSIZE] = (_Float16)o;
  }
}

// ---------- kernel 2: logits[bh][m][n] = (q.k)/8, masked
__global__ __launch_bounds__(256) void attn_logits(
    const _Float16* __restrict__ qbuf,
    const _Float16* __restrict__ kbuf,
    const int*      __restrict__ am,    // (8, 512) int32
    float*          __restrict__ out) { // (8,12,512,512)
  const int wave = threadIdx.x >> 5;
  const int lane = threadIdx.x & 31;
  const int tile = blockIdx.x * 8 + wave;      // 0 .. 96*32*32-1
  const int bh  = tile >> 10;
  const int rem = tile & 1023;
  const int mt  = rem >> 5;
  const int nt  = rem & 31;
  const int j   = lane & 15;
  const int hi  = lane >> 4;
  const int m0  = mt * 16, n0 = nt * 16;

  float* obase = out + ((size_t)bh * SEQ + m0) * SEQ + n0;

  if (nt < mt) {                               // fully below diagonal -> -inf
    #pragma unroll
    for (int r = 0; r < 8; ++r)
      obase[(r + 8 * hi) * SEQ + j] = NEGINF;
    return;
  }

  const _Float16* qp = qbuf + ((bh * SEQ) + m0 + j) * HSIZE + 8 * hi;
  const _Float16* kp = kbuf + ((bh * SEQ) + n0 + j) * HSIZE + 8 * hi;

  v8f acc = {};
  #pragma unroll
  for (int dc = 0; dc < 2; ++dc) {             // HSIZE=64 -> 2 chunks of K=32
    v8h a0 = *(const v8h*)(qp + dc * 32);
    v8h a1 = *(const v8h*)(qp + dc * 32 + 16);
    v8h b0 = *(const v8h*)(kp + dc * 32);
    v8h b1 = *(const v8h*)(kp + dc * 32 + 16);
    acc = __builtin_amdgcn_wmma_f32_16x16x32_f16(
        false, cat16(a0, a1), false, cat16(b0, b1),
        (short)0, acc, false, false);
  }

  const int b = bh / HEADS;
  const int n = n0 + j;
  const int cmask = am[b * SEQ + n];
  #pragma unroll
  for (int r = 0; r < 8; ++r) {
    int m = m0 + r + 8 * hi;
    int rmask = am[b * SEQ + m];
    bool valid = (cmask != 0) && (rmask != 0) && (n >= m);
    obase[(r + 8 * hi) * SEQ + j] = valid ? acc[r] * 0.125f : NEGINF;
  }
}

extern "C" void kernel_launch(void* const* d_in, const int* in_sizes, int n_in,
                              void* d_out, int out_size, void* d_ws, size_t ws_size,
                              hipStream_t stream) {
  const float* X    = (const float*)d_in[0];   // (8,512,768)
  const float* W    = (const float*)d_in[1];   // (768,1536)
  const float* bias = (const float*)d_in[2];   // (1536,)
  const int*   am   = (const int*)d_in[3];     // (8,512)
  float* out = (float*)d_out;

  char* ws = (char*)d_ws;
  // qbuf: 6291456 B ; kbuf: 6291456 B ; Wt: 2359296 B ; Xh: 6291456 B ;
  // sintab/costab: 65536 B each  -> total ~21.4 MB
  _Float16* qbuf = (_Float16*)(ws);
  _Float16* kbuf = (_Float16*)(ws + 6291456);
  _Float16* Wt   = (_Float16*)(ws + 12582912);
  _Float16* Xh   = (_Float16*)(ws + 14942208);
  float* sintab  = (float*)(ws + 21233664);
  float* costab  = (float*)(ws + 21299200);

  pack_w<<<(OUTDIM * HIDDEN + 255) / 256, 256, 0, stream>>>(W, Wt);
  pack_x<<<(NTOK * HIDDEN / 4 + 255) / 256, 256, 0, stream>>>(X, Xh);
  rope_tables<<<(NPAIR * SEQ + 255) / 256, 256, 0, stream>>>(sintab, costab);
  // 256 token-tiles x 96 channel-tiles, 8 waves (tiles) per block
  qk_proj_rope<<<(256 * 96) / 8, 256, 0, stream>>>(Xh, Wt, bias, sintab, costab,
                                                   qbuf, kbuf);
  // 96 bh x 32 x 32 tiles, 8 waves per block
  attn_logits<<<(96 * 32 * 32) / 8, 256, 0, stream>>>(qbuf, kbuf, am, out);
}
